// FullAttention_17300128268955
// MI455X (gfx1250) — matrix operand order repaired and checked by reading, not compiled, not run on hardware
//
#include <hip/hip_runtime.h>

// ---------------- problem constants ----------------
constexpr int B_ = 2;
constexpr int T_ = 2048;
constexpr int C_ = 1024;
constexpr int H_ = 16;
constexpr int D_ = 64;

typedef __attribute__((ext_vector_type(16))) _Float16 v16h;
typedef __attribute__((ext_vector_type(8)))  float    v8f;
typedef __attribute__((ext_vector_type(4)))  unsigned int u32x4;

union F16x16 {
    v16h v;
    _Float16 h[16];
    unsigned int u[8];
    u32x4 q[2];
};

#define NEG_INF (-__builtin_inff())

__device__ inline v8f wmma_f16(v16h a, v16h b, v8f c) {
    // D = A(16x32 f16) * B(32x16 f16) + C(16x16 f32)
    return __builtin_amdgcn_wmma_f32_16x16x32_f16(
        /*neg_a=*/false, a, /*neg_b=*/false, b,
        /*c_mod=*/(short)0, c, /*reuse_a=*/false, /*reuse_b=*/false);
}

// CDNA5 async global->LDS copy (ASYNCcnt-tracked); LDS address is the low
// 32 bits of the generic shared-memory pointer (flat LDS aperture truncation).
__device__ inline void async_b128(const _Float16* g, _Float16* l) {
    unsigned int loff = (unsigned int)(unsigned long long)(uintptr_t)l;
    asm volatile("global_load_async_to_lds_b128 %0, %1, off"
                 :: "v"(loff), "v"(g)
                 : "memory");
}
__device__ inline void wait_async0() {
    asm volatile("s_wait_asynccnt 0" ::: "memory");
}

__device__ inline float rowmax16(float v) {
    v = fmaxf(v, __shfl_xor(v, 1, 32));
    v = fmaxf(v, __shfl_xor(v, 2, 32));
    v = fmaxf(v, __shfl_xor(v, 4, 32));
    v = fmaxf(v, __shfl_xor(v, 8, 32));
    return v;
}
__device__ inline float rowsum16(float v) {
    v += __shfl_xor(v, 1, 32);
    v += __shfl_xor(v, 2, 32);
    v += __shfl_xor(v, 4, 32);
    v += __shfl_xor(v, 8, 32);
    return v;
}

// ---------------- f32 -> f16 pack ----------------
__global__ void cvt_f32_f16_kernel(const float* __restrict__ in,
                                   _Float16* __restrict__ out, int n) {
    int i = blockIdx.x * blockDim.x + threadIdx.x;
    if (i < n) out[i] = (_Float16)in[i];
}

// ---------------- tiled WMMA GEMM: C[M,N] = A[M,K]*B[K,N] + bias ----------------
// block 256 threads (8 waves); tile 128x128, K-step 32; wave w owns rows
// w*16..w*16+15 across all 128 cols (8 WMMA tiles, one shared A fragment).
// Double-buffered LDS filled by async global->LDS copies. A sched_barrier pins
// all A/B fragment ds_loads before the WMMA chain so the backend emits partial
// dscnt waits (loads for later fragments stay in flight during earlier WMMAs).
#define GM 128
#define GN 128
#define GK 32
#define LDA_ (GK + 8)           // 40 halves  (A row stride)
#define LDB_ (GN + 8)           // 136 halves (B row stride)
#define ABUF (GM * LDA_)        // 5120 halves
#define SBUF (ABUF + GK * LDB_) // 5120 + 4352 = 9472 halves per buffer

__global__ __launch_bounds__(256)
void gemm_f16_kernel(const _Float16* __restrict__ A,   // [M,K] row-major (f16)
                     const _Float16* __restrict__ Bm,  // [K,N] row-major (f16)
                     const float* __restrict__ bias,   // [N]
                     float* __restrict__ C,            // [M,N] f32
                     int M, int N, int K) {
    __shared__ __align__(16) _Float16 smem[2][SBUF];

    const int tid  = threadIdx.x;
    const int wave = tid >> 5;
    const int lane = tid & 31;
    const int lm = lane & 15, hi = lane >> 4;
    const int m0 = blockIdx.y * GM;
    const int n0 = blockIdx.x * GN;

    // staging coordinates: A = 2 chunks of 8 halves/thread, B = 2 chunks/thread
    const int ar = tid >> 2,  ac8 = (tid & 3) * 8;    // rows 0..63 (+64 second)
    const int br = tid >> 4,  bc8 = (tid & 15) * 8;   // rows 0..15 (+16 second)
    const _Float16* gA = A  + (size_t)(m0 + ar) * K + ac8;
    const _Float16* gB = Bm + (size_t)br * N + n0 + bc8;

    _Float16* dA0[2]; _Float16* dA1[2]; _Float16* dB0[2]; _Float16* dB1[2];
#pragma unroll
    for (int u = 0; u < 2; ++u) {
        dA0[u] = &smem[u][ar * LDA_ + ac8];
        dA1[u] = &smem[u][(ar + 64) * LDA_ + ac8];
        dB0[u] = &smem[u][ABUF + br * LDB_ + bc8];
        dB1[u] = &smem[u][ABUF + (br + 16) * LDB_ + bc8];
    }

    v8f acc[8] = {};

    auto issue = [&](int ki, int buf) {
        const _Float16* a = gA + (size_t)ki * GK;
        const _Float16* b = gB + (size_t)ki * GK * N;
        async_b128(a, dA0[buf]);
        async_b128(a + (size_t)64 * K, dA1[buf]);
        async_b128(b, dB0[buf]);
        async_b128(b + (size_t)16 * N, dB1[buf]);
    };

    const int nk = K / GK;
    issue(0, 0);
    wait_async0();
    __syncthreads();

    for (int ki = 0; ki < nk; ++ki) {
        const int cur = ki & 1, nxt = cur ^ 1;
        if (ki + 1 < nk) issue(ki + 1, nxt);   // prefetch next tile (other buffer)

        // A fragment (16x32 f16): lane lm = row; halves per ISA layout
        F16x16 a;
#pragma unroll
        for (int p = 0; p < 8; ++p) {
            int kk = ((p >= 4) ? 16 : 0) + hi * 8 + (p & 3) * 2;
            a.u[p] = *(const unsigned int*)&smem[cur][(wave * 16 + lm) * LDA_ + kk];
        }
        // preload all 8 B fragments (32x16 each): lane holds column n = lm, k in table order
        F16x16 bfr[8];
#pragma unroll
        for (int c = 0; c < 8; ++c) {
#pragma unroll
            for (int s = 0; s < 16; ++s) {
                int kb = ((s >= 8) ? 16 : 0) + hi * 8 + (s & 7);
                bfr[c].h[s] = smem[cur][ABUF + kb * LDB_ + c * 16 + lm];
            }
        }
        // pin schedule: all ds_loads above, all WMMAs below -> partial dscnt waits
        __builtin_amdgcn_sched_barrier(0);
        // 8 back-to-back WMMAs (independent accumulators, shared A)
#pragma unroll
        for (int c = 0; c < 8; ++c)
            acc[c] = wmma_f16(a.v, bfr[c].v, acc[c]);

        if (ki + 1 < nk) wait_async0();        // own async writes to nxt landed
        __syncthreads();                        // all waves: writes visible, reads done
    }

    // epilogue: C/D layout — lane col = lm, rows = hi*8 + r
#pragma unroll
    for (int c = 0; c < 8; ++c) {
        int n = n0 + c * 16 + lm;
        float bval = bias[n];
#pragma unroll
        for (int r = 0; r < 8; ++r) {
            int m = m0 + wave * 16 + hi * 8 + r;
            C[(size_t)m * N + n] = acc[c][r] + bval;
        }
    }
}

// ---------------- RMSNorm + repack to head layout ----------------
// mode 0: q -> [B*H, T, D], scaled by outscale (1/sqrt(D))
// mode 1: k -> [B*H, D, T] (transposed for K^T B-fragments)
// mode 2: v -> [B*H, T, D], no norm
__global__ __launch_bounds__(256)
void rmsnorm_pack_kernel(const float* __restrict__ raw,  // [B*T, C]
                         const float* __restrict__ g,    // [C] or nullptr
                         _Float16* __restrict__ out,
                         int mode, float outscale) {
    __shared__ float red[256];
    const int token = blockIdx.x;          // b*T + t
    const int b = token / T_, t = token % T_;
    const int tid = threadIdx.x;

    float vals[4];
    float ss = 0.0f;
#pragma unroll
    for (int i = 0; i < 4; ++i) {
        vals[i] = raw[(size_t)token * C_ + tid + i * 256];
        ss += vals[i] * vals[i];
    }
    red[tid] = ss;
    for (int s = 128; s > 0; s >>= 1) {
        __syncthreads();
        if (tid < s) red[tid] += red[tid + s];
    }
    __syncthreads();
    float rinv = (mode == 2) ? 1.0f : rsqrtf(red[0] * (1.0f / C_) + 1e-6f);

#pragma unroll
    for (int i = 0; i < 4; ++i) {
        int idx = tid + i * 256;
        int h = idx >> 6, d = idx & 63;
        float o = vals[i] * rinv * outscale * (g ? g[idx] : 1.0f);
        size_t dst;
        if (mode == 1)
            dst = ((size_t)(b * H_ + h) * D_ + d) * T_ + t;
        else
            dst = ((size_t)(b * H_ + h) * T_ + t) * D_ + d;
        out[dst] = (_Float16)o;
    }
}

// ---------------- flash attention forward ----------------
// grid (T/64, B*H), block 128 (4 waves). wave owns 16 query rows resident as
// A-fragments; streams 32 keys/iter: QK^T via WMMA, online softmax, P staged
// through LDS (C-layout -> A-layout), PV via WMMA. V fragment loads are
// pinned (sched_barrier) before the softmax VALU block to hide their latency.
__global__ __launch_bounds__(128)
void flash_fwd_kernel(const _Float16* __restrict__ qh,   // [B*H, T, D] (pre-scaled)
                      const _Float16* __restrict__ kT,   // [B*H, D, T]
                      const _Float16* __restrict__ vh,   // [B*H, T, D]
                      const unsigned char* __restrict__ mask,  // [B, T]
                      _Float16* __restrict__ yh,         // [B, T, C]
                      float* __restrict__ smax,          // [B*H, T]
                      float* __restrict__ ssum) {        // [B*H, T]
    __shared__ __align__(16) _Float16 ldsP[4][16 * 34];

    const int wave = threadIdx.x >> 5, lane = threadIdx.x & 31;
    const int lm = lane & 15, hi = lane >> 4;
    const int bh = blockIdx.y;
    const int b = bh / H_, h = bh % H_;
    const int mrow0 = blockIdx.x * 64 + wave * 16;

    const _Float16* qp = qh + (size_t)bh * T_ * D_;
    const _Float16* kp = kT + (size_t)bh * D_ * T_;
    const _Float16* vp = vh + (size_t)bh * T_ * D_;
    const unsigned char* mp = mask + (size_t)b * T_;

    // resident Q fragments: d-chunks [0,32) and [32,64)
    F16x16 aq0, aq1;
    {
        const _Float16* qrow = qp + (size_t)(mrow0 + lm) * D_;
#pragma unroll
        for (int p = 0; p < 8; ++p) {
            int d = ((p >= 4) ? 16 : 0) + hi * 8 + (p & 3) * 2;
            aq0.u[p] = *(const unsigned int*)(qrow + d);
            aq1.u[p] = *(const unsigned int*)(qrow + 32 + d);
        }
    }

    float ml[8], ll[8];
#pragma unroll
    for (int r = 0; r < 8; ++r) { ml[r] = NEG_INF; ll[r] = 0.0f; }
    v8f accO[4] = {};

    for (int j0 = 0; j0 < T_; j0 += 32) {
        // K^T B-fragments: B[dk][j] = kT[d][j]; lane column j = lm, dk in table order
        F16x16 b00, b01, b10, b11;
#pragma unroll
        for (int s = 0; s < 16; ++s) {
            int dk = ((s >= 8) ? 16 : 0) + hi * 8 + (s & 7);
            const _Float16* kr0 = kp + (size_t)dk * T_ + j0;         // d chunk 0
            const _Float16* kr1 = kp + (size_t)(32 + dk) * T_ + j0;  // d chunk 1
            b00.h[s] = kr0[lm];  b10.h[s] = kr0[16 + lm];
            b01.h[s] = kr1[lm];  b11.h[s] = kr1[16 + lm];
        }

        v8f s0 = {}, s1 = {};
        s0 = wmma_f16(aq0.v, b00.v, s0);
        s0 = wmma_f16(aq1.v, b01.v, s0);
        s1 = wmma_f16(aq0.v, b10.v, s1);
        s1 = wmma_f16(aq1.v, b11.v, s1);

        // V B-fragments issued early: independent of S; sched_barrier pins the
        // issue point so softmax VALU/transcendentals hide their latency.
        F16x16 vfr[4];
#pragma unroll
        for (int s = 0; s < 16; ++s) {
            int jk = ((s >= 8) ? 16 : 0) + hi * 8 + (s & 7);
            const _Float16* vr = vp + (size_t)(j0 + jk) * D_;
#pragma unroll
            for (int c = 0; c < 4; ++c)
                vfr[c].h[s] = vr[c * 16 + lm];
        }
        __builtin_amdgcn_sched_barrier(0);

        const bool km0 = mp[j0 + lm] != 0;
        const bool km1 = mp[j0 + 16 + lm] != 0;

#pragma unroll
        for (int r = 0; r < 8; ++r) {
            float c0 = km0 ? s0[r] : NEG_INF;
            float c1 = km1 ? s1[r] : NEG_INF;
            float bm = rowmax16(fmaxf(c0, c1));
            float mnew = fmaxf(ml[r], bm);
            float meff = (mnew == NEG_INF) ? 0.0f : mnew;
            float alpha = (ml[r] == NEG_INF) ? 0.0f : __expf(ml[r] - mnew);
            float p0 = km0 ? __expf(s0[r] - meff) : 0.0f;
            float p1 = km1 ? __expf(s1[r] - meff) : 0.0f;
            float rs = rowsum16(p0 + p1);
            ll[r] = ll[r] * alpha + rs;
            ml[r] = mnew;
#pragma unroll
            for (int c = 0; c < 4; ++c) accO[c][r] *= alpha;
            int row = hi * 8 + r;
            ldsP[wave][row * 34 + lm]      = (_Float16)p0;
            ldsP[wave][row * 34 + 16 + lm] = (_Float16)p1;
        }

        // reload P as 16x32 A-fragment (per-wave LDS; DS ops in-order per wave)
        F16x16 aP;
#pragma unroll
        for (int p = 0; p < 8; ++p) {
            int kk = ((p >= 4) ? 16 : 0) + hi * 8 + (p & 3) * 2;
            aP.u[p] = *(const unsigned int*)&ldsP[wave][lm * 34 + kk];
        }

#pragma unroll
        for (int c = 0; c < 4; ++c)
            accO[c] = wmma_f16(aP.v, vfr[c].v, accO[c]);
    }

    // normalize + write yh (f16, [B,T,C] interleaved head layout) + stats
#pragma unroll
    for (int r = 0; r < 8; ++r) {
        int t = mrow0 + hi * 8 + r;
        float inv = (ll[r] > 0.0f) ? 1.0f / ll[r] : 0.0f;
#pragma unroll
        for (int c = 0; c < 4; ++c)
            yh[((size_t)(b * T_ + t)) * C_ + h * D_ + c * 16 + lm] =
                (_Float16)(accO[c][r] * inv);
        if (lm == 0) {
            smax[(size_t)bh * T_ + t] = ml[r];
            ssum[(size_t)bh * T_ + t] = ll[r];
        }
    }
}

// ---------------- att_mean: recompute S, apply stats, average heads ----------------
// grid (T/32, T/64, B), block 128 (4 waves); wave = 16 rows x 32 cols, loops heads.
__global__ __launch_bounds__(128)
void att_mean_kernel(const _Float16* __restrict__ qh,   // [B*H, T, D]
                     const _Float16* __restrict__ kT,   // [B*H, D, T]
                     const unsigned char* __restrict__ mask,
                     const float* __restrict__ smax,
                     const float* __restrict__ ssum,
                     float* __restrict__ out) {         // [B, T, T]
    const int wave = threadIdx.x >> 5, lane = threadIdx.x & 31;
    const int lm = lane & 15, hi = lane >> 4;
    const int b = blockIdx.z;
    const int mrow0 = blockIdx.y * 64 + wave * 16;
    const int j0 = blockIdx.x * 32;

    const bool km0 = mask[(size_t)b * T_ + j0 + lm] != 0;
    const bool km1 = mask[(size_t)b * T_ + j0 + 16 + lm] != 0;

    float acc0[8] = {}, acc1[8] = {};
    const int dk = lm + 16 * hi;

    for (int h = 0; h < H_; ++h) {
        const int bh = b * H_ + h;
        const _Float16* qp = qh + (size_t)bh * T_ * D_;
        const _Float16* kp = kT + (size_t)bh * D_ * T_;

        F16x16 aq0, aq1;
        const _Float16* qrow = qp + (size_t)(mrow0 + lm) * D_;
#pragma unroll
        for (int p = 0; p < 8; ++p) {
            int d = ((p >= 4) ? 16 : 0) + hi * 8 + (p & 3) * 2;
            aq0.u[p] = *(const unsigned int*)(qrow + d);
            aq1.u[p] = *(const unsigned int*)(qrow + 32 + d);
        }
        F16x16 b00, b01, b10, b11;
#pragma unroll
        for (int s = 0; s < 16; ++s) {
            int dks = ((s >= 8) ? 16 : 0) + hi * 8 + (s & 7);
            const _Float16* kr0 = kp + (size_t)dks * T_ + j0;
            const _Float16* kr1 = kp + (size_t)(32 + dks) * T_ + j0;
            b00.h[s] = kr0[lm];  b10.h[s] = kr0[16 + lm];
            b01.h[s] = kr1[lm];  b11.h[s] = kr1[16 + lm];
        }

        v8f s0 = {}, s1 = {};
        s0 = wmma_f16(aq0.v, b00.v, s0);
        s0 = wmma_f16(aq1.v, b01.v, s0);
        s1 = wmma_f16(aq0.v, b10.v, s1);
        s1 = wmma_f16(aq1.v, b11.v, s1);

#pragma unroll
        for (int r = 0; r < 8; ++r) {
            int t = mrow0 + hi * 8 + r;
            float m = smax[(size_t)bh * T_ + t];
            float l = ssum[(size_t)bh * T_ + t];
            float meff = (m == NEG_INF) ? 0.0f : m;
            float inv = (l > 0.0f) ? 1.0f / l : 0.0f;
            acc0[r] += km0 ? __expf(s0[r] - meff) * inv : 0.0f;
            acc1[r] += km1 ? __expf(s1[r] - meff) * inv : 0.0f;
        }
    }

    const float invH = 1.0f / (float)H_;
#pragma unroll
    for (int r = 0; r < 8; ++r) {
        int t = mrow0 + hi * 8 + r;
        out[((size_t)b * T_ + t) * T_ + j0 + lm]      = acc0[r] * invH;
        out[((size_t)b * T_ + t) * T_ + j0 + 16 + lm] = acc1[r] * invH;
    }
}

// ---------------- launcher ----------------
extern "C" void kernel_launch(void* const* d_in, const int* in_sizes, int n_in,
                              void* d_out, int out_size, void* d_ws, size_t ws_size,
                              hipStream_t stream) {
    (void)in_sizes; (void)n_in; (void)out_size; (void)ws_size;

    const float* x  = (const float*)d_in[0];
    const unsigned char* mask = (const unsigned char*)d_in[1];
    const float* Wq = (const float*)d_in[2];
    const float* bq = (const float*)d_in[3];
    const float* Wk = (const float*)d_in[4];
    const float* bk = (const float*)d_in[5];
    const float* Wv = (const float*)d_in[6];
    const float* bv = (const float*)d_in[7];
    const float* gq = (const float*)d_in[8];
    const float* gk = (const float*)d_in[9];
    const float* Wp = (const float*)d_in[10];
    const float* bp = (const float*)d_in[11];

    float* y_out   = (float*)d_out;                     // [B,T,C]
    float* att_out = y_out + (size_t)B_ * T_ * C_;      // [B,T,T]

    char* ws = (char*)d_ws;
    size_t off = 0;
    auto take = [&](size_t bytes) -> char* {
        char* p = ws + off;
        off = (off + bytes + 255) & ~(size_t)255;
        return p;
    };
    const size_t NTOK = (size_t)B_ * T_;
    _Float16* xh   = (_Float16*)take(NTOK * C_ * 2);
    _Float16* Wqh  = (_Float16*)take((size_t)C_ * C_ * 2);
    _Float16* Wkh  = (_Float16*)take((size_t)C_ * C_ * 2);
    _Float16* Wvh  = (_Float16*)take((size_t)C_ * C_ * 2);
    _Float16* Wph  = (_Float16*)take((size_t)C_ * C_ * 2);
    _Float16* qh   = (_Float16*)take(NTOK * C_ * 2);   // [B*H, T, D]
    _Float16* kTt  = (_Float16*)take(NTOK * C_ * 2);   // [B*H, D, T]
    _Float16* vh   = (_Float16*)take(NTOK * C_ * 2);   // [B*H, T, D]
    _Float16* yh   = (_Float16*)take(NTOK * C_ * 2);   // [B, T, C]
    float*    raw  = (float*)take(NTOK * C_ * 4);      // reused q/k/v raw
    float*    smax = (float*)take((size_t)B_ * H_ * T_ * 4);
    float*    ssum = (float*)take((size_t)B_ * H_ * T_ * 4);

    // pack f32 -> f16
    {
        int nx = (int)(NTOK * C_);
        cvt_f32_f16_kernel<<<(nx + 255) / 256, 256, 0, stream>>>(x, xh, nx);
        int nw = C_ * C_;
        cvt_f32_f16_kernel<<<(nw + 255) / 256, 256, 0, stream>>>(Wq, Wqh, nw);
        cvt_f32_f16_kernel<<<(nw + 255) / 256, 256, 0, stream>>>(Wk, Wkh, nw);
        cvt_f32_f16_kernel<<<(nw + 255) / 256, 256, 0, stream>>>(Wv, Wvh, nw);
        cvt_f32_f16_kernel<<<(nw + 255) / 256, 256, 0, stream>>>(Wp, Wph, nw);
    }

    const dim3 gemm_grid(C_ / GN, (int)(NTOK / GM));
    const float qscale = 0.125f;  // 1/sqrt(D)

    // Q = rmsnorm(x@Wq + bq) * gq * qscale
    gemm_f16_kernel<<<gemm_grid, 256, 0, stream>>>(xh, Wqh, bq, raw,
                                                   (int)NTOK, C_, C_);
    rmsnorm_pack_kernel<<<(int)NTOK, 256, 0, stream>>>(raw, gq, qh, 0, qscale);
    // K (transposed layout)
    gemm_f16_kernel<<<gemm_grid, 256, 0, stream>>>(xh, Wkh, bk, raw,
                                                   (int)NTOK, C_, C_);
    rmsnorm_pack_kernel<<<(int)NTOK, 256, 0, stream>>>(raw, gk, kTt, 1, 1.0f);
    // V (no norm)
    gemm_f16_kernel<<<gemm_grid, 256, 0, stream>>>(xh, Wvh, bv, raw,
                                                   (int)NTOK, C_, C_);
    rmsnorm_pack_kernel<<<(int)NTOK, 256, 0, stream>>>(raw, nullptr, vh, 2, 1.0f);

    // flash attention -> yh + stats
    flash_fwd_kernel<<<dim3(T_ / 64, B_ * H_), 128, 0, stream>>>(
        qh, kTt, vh, mask, yh, smax, ssum);

    // att_mean -> second output region
    att_mean_kernel<<<dim3(T_ / 32, T_ / 64, B_), 128, 0, stream>>>(
        qh, kTt, mask, smax, ssum, att_out);

    // output projection -> first output region
    gemm_f16_kernel<<<gemm_grid, 256, 0, stream>>>(yh, Wph, bp, y_out,
                                                   (int)NTOK, C_, C_);
}
